// CifPredictorV2_23648089932142
// MI455X (gfx1250) — matrix operand order repaired and checked by reading, not compiled, not run on hardware
//
#include <hip/hip_runtime.h>
#include <math.h>

typedef __bf16 bf16_t;
typedef __attribute__((ext_vector_type(16))) __bf16 v16bf;
typedef __attribute__((ext_vector_type(8)))  __bf16 v8bf;
typedef __attribute__((ext_vector_type(8)))  float  v8f;

#define B_  16
#define T_  2048
#define H_  512
#define KW  3
#define BM  64                 // t-rows per block
#define LROWS (BM + 2)         // +2 halo rows for conv taps
#define LSTR  (H_ + 8)         // padded LDS row stride (bank-conflict avoidance)

// ---- workspace layout (bytes) ----
#define WT_BYTES  (KW * H_ * H_ * 2)            // bf16 repacked weights: 1,572,864
#define WS_WT     0
#define WS_ALPHA  (WS_WT + WT_BYTES)            // float[B*T]
#define WS_CUR    (WS_ALPHA + B_*T_*4)
#define WS_REM    (WS_CUR   + B_*T_*4)
#define WS_ROWA   (WS_REM   + B_*T_*4)
#define WS_ROWB   (WS_ROWA  + B_*T_*4)
#define WS_NF     (WS_ROWB  + B_*T_*4)          // int[B]

// ---------------------------------------------------------------------------
// Repack conv_w [O][I][3] f32  ->  Wt [k][O][I] bf16 (contiguous along I = K)
// ---------------------------------------------------------------------------
__global__ void pack_w_kernel(const float* __restrict__ conv_w,
                              bf16_t* __restrict__ Wt) {
  int idx = blockIdx.x * blockDim.x + threadIdx.x;
  int total = KW * H_ * H_;
  if (idx < total) {
    int k = idx / (H_ * H_);
    int r = idx % (H_ * H_);
    int o = r / H_;
    int i = r % H_;
    Wt[idx] = (bf16_t)conv_w[(o * H_ + i) * KW + k];
  }
}

// ---------------------------------------------------------------------------
// Zero the output (harness poisons it; scatter uses atomics, so must zero)
// ---------------------------------------------------------------------------
__global__ void zero_out_kernel(float4* __restrict__ out, int n4) {
  int stride = gridDim.x * blockDim.x;
  for (int i = blockIdx.x * blockDim.x + threadIdx.x; i < n4; i += stride)
    out[i] = make_float4(0.f, 0.f, 0.f, 0.f);
}

// ---------------------------------------------------------------------------
// alphas GEMM: mem = conv(hidden) ; out = relu(mem + conv_b + hidden)
//              alpha = sigmoid(out . out_w + out_b) * mask
// Block: 64 t-rows x 512 outputs, 8 waves (each wave: 4 M-tiles x 4 N-tiles)
// ---------------------------------------------------------------------------
__global__ __launch_bounds__(256)
void alphas_gemm_kernel(const float* __restrict__ hidden,
                        const float* __restrict__ mask,
                        const bf16_t* __restrict__ Wt,
                        const float* __restrict__ conv_b,
                        const float* __restrict__ out_w,
                        const float* __restrict__ out_b,
                        float* __restrict__ alphas) {
  extern __shared__ char smem[];
  bf16_t* sA   = (bf16_t*)smem;                       // [LROWS][LSTR] bf16
  float*  sRed = (float*)(smem + LROWS * LSTR * 2);   // [BM][8]

  const int b   = blockIdx.y;
  const int t0  = blockIdx.x * BM;
  const int tid = threadIdx.x;
  const int lane = tid & 31;
  const int wv   = tid >> 5;          // 0..7 -> N slice of 64
  const int o0   = wv * 64;
  const int halfsel = lane >> 4;      // 0: lanes 0-15, 1: lanes 16-31
  const int lm      = lane & 15;

  // ---- stage A tile (rows t0-1 .. t0+64) into LDS as bf16 ----
  for (int e = tid; e < LROWS * (H_ / 4); e += 256) {
    int r  = e / (H_ / 4);
    int c4 = (e % (H_ / 4)) * 4;
    int t  = t0 + r - 1;
    float4 v = make_float4(0.f, 0.f, 0.f, 0.f);
    if (t >= 0 && t < T_)
      v = *(const float4*)(hidden + ((size_t)b * T_ + t) * H_ + c4);
    bf16_t* dst = sA + r * LSTR + c4;
    dst[0] = (bf16_t)v.x; dst[1] = (bf16_t)v.y;
    dst[2] = (bf16_t)v.z; dst[3] = (bf16_t)v.w;
  }
  __syncthreads();

  v8f acc[4][4] = {};   // [Mtile][Ntile] f32 accumulators

  // ---- K loop: 3 conv taps x 512 channels, step 32 ----
  for (int k = 0; k < KW; ++k) {
    const bf16_t* Wk = Wt + (size_t)k * H_ * H_;
    for (int i0 = 0; i0 < H_; i0 += 32) {
      // A fragments: lane<16 -> K {i0..i0+15}; lane>=16 -> K {i0+16..i0+31}
      const int aoff = i0 + halfsel * 16;
      v16bf afrag[4];
#pragma unroll
      for (int mt = 0; mt < 4; ++mt) {
        const bf16_t* ap = sA + (mt * 16 + lm + k) * LSTR + aoff;
        v8bf lo = *(const v8bf*)ap;
        v8bf hi = *(const v8bf*)(ap + 8);
        v16bf a;
#pragma unroll
        for (int j = 0; j < 8; ++j) { a[j] = lo[j]; a[j + 8] = hi[j]; }
        afrag[mt] = a;
      }
      // B fragments: column N = lm, K contiguous 16 per lane-half
      const int boff = i0 + halfsel * 16;
#pragma unroll
      for (int nt = 0; nt < 4; ++nt) {
        const int o = o0 + nt * 16 + lm;
        const bf16_t* bp = Wk + (size_t)o * H_ + boff;
        __builtin_prefetch(bp + 32, 0, 1);     // next K-step (global_prefetch_b8)
        v16bf bfrag = *(const v16bf*)bp;
#pragma unroll
        for (int mt = 0; mt < 4; ++mt) {
          acc[mt][nt] = __builtin_amdgcn_wmma_f32_16x16x32_bf16(
              false, afrag[mt], false, bfrag, (short)0, acc[mt][nt],
              false, false);
        }
      }
    }
  }

  // ---- epilogue: residual + bias + relu, weighted row-sums for the dot ----
  // D layout: lane<16: N=lm, M=r (VGPR r); lane>=16: N=lm, M=r+8
  float rs[4][8];
#pragma unroll
  for (int mt = 0; mt < 4; ++mt)
#pragma unroll
    for (int r = 0; r < 8; ++r) rs[mt][r] = 0.f;

#pragma unroll
  for (int nt = 0; nt < 4; ++nt) {
    const int o   = o0 + nt * 16 + lm;
    const float owv = out_w[o];
    const float cbv = conv_b[o];
#pragma unroll
    for (int mt = 0; mt < 4; ++mt) {
#pragma unroll
      for (int r = 0; r < 8; ++r) {
        const int m = mt * 16 + halfsel * 8 + r;
        const float hv = hidden[((size_t)b * T_ + t0 + m) * H_ + o];
        float v = acc[mt][nt][r] + cbv + hv;
        v = fmaxf(v, 0.f);
        rs[mt][r] += v * owv;
      }
    }
  }
  // reduce over the 16 lanes of each half-wave (same rows, different o)
#pragma unroll
  for (int off = 1; off < 16; off <<= 1)
#pragma unroll
    for (int mt = 0; mt < 4; ++mt)
#pragma unroll
      for (int r = 0; r < 8; ++r)
        rs[mt][r] += __shfl_xor(rs[mt][r], off, 32);

  if (lm == 0) {
#pragma unroll
    for (int mt = 0; mt < 4; ++mt)
#pragma unroll
      for (int r = 0; r < 8; ++r)
        sRed[(mt * 16 + halfsel * 8 + r) * 8 + wv] = rs[mt][r];
  }
  __syncthreads();

  if (tid < BM) {
    float s = out_b[0];
#pragma unroll
    for (int w = 0; w < 8; ++w) s += sRed[tid * 8 + w];
    float sig = 1.f / (1.f + expf(-s));
    float a = fmaxf(sig * 1.0f - 0.0f, 0.f);          // SMOOTH=1, NOISE=0
    a *= mask[(size_t)b * T_ + t0 + tid];
    alphas[(size_t)b * T_ + t0 + tid] = a;
  }
}

// ---------------------------------------------------------------------------
// CIF scan: prefix-sum alphas per batch; derive fire rows + weights.
// alpha in (0,1) => fires == integer crossings of the prefix sum (>= ties ok).
// ---------------------------------------------------------------------------
__global__ __launch_bounds__(256)
void cif_scan_kernel(const float* __restrict__ alphas,
                     float* __restrict__ curW, float* __restrict__ remW,
                     int* __restrict__ rowA, int* __restrict__ rowB,
                     int* __restrict__ nF) {
  __shared__ float part[256];
  const int b = blockIdx.x;
  const int tid = threadIdx.x;
  const float* al = alphas + (size_t)b * T_;
  const int PER = T_ / 256;   // 8

  float loc[PER];
  float s = 0.f;
#pragma unroll
  for (int j = 0; j < PER; ++j) { loc[j] = al[tid * PER + j]; s += loc[j]; }
  part[tid] = s;
  __syncthreads();
  for (int off = 1; off < 256; off <<= 1) {
    float v = (tid >= off) ? part[tid - off] : 0.f;
    __syncthreads();
    part[tid] += v;
    __syncthreads();
  }
  float S = (tid == 0) ? 0.f : part[tid - 1];   // exclusive offset

#pragma unroll
  for (int j = 0; j < PER; ++j) {
    const float a = loc[j];
    const float Sprev = S;
    S += a;
    const float Fp = floorf(Sprev);
    const float Fc = floorf(S);
    const int fire = (Fc > Fp);
    const float cur = fire ? (1.0f - (Sprev - Fp)) : a;
    const float rem = fire ? (a - cur) : 0.f;
    const size_t t = (size_t)b * T_ + tid * PER + j;
    curW[t] = cur;
    remW[t] = rem;
    rowA[t] = (int)Fp;
    rowB[t] = (int)Fc;
  }
  if (tid == 255) nF[b] = (int)floorf(part[255]);
}

// ---------------------------------------------------------------------------
// Scatter: acoustic[b,row] += w * hidden[b,t]  (rows closed by a later fire)
// ---------------------------------------------------------------------------
__global__ __launch_bounds__(128)
void cif_scatter_kernel(const float* __restrict__ hidden,
                        const float* __restrict__ curW,
                        const float* __restrict__ remW,
                        const int* __restrict__ rowA,
                        const int* __restrict__ rowB,
                        const int* __restrict__ nF,
                        float* __restrict__ out) {
  const int t = blockIdx.x;
  const int b = blockIdx.y;
  const size_t bt = (size_t)b * T_ + t;
  const float cur = curW[bt];
  const float rem = remW[bt];
  const int ra = rowA[bt];
  const int rb = rowB[bt];
  const int n  = nF[b];
  const int h = threadIdx.x * 4;

  float4 hv = *(const float4*)(hidden + bt * H_ + h);

  if (ra < n) {
    float* p = out + ((size_t)b * T_ + ra) * H_ + h;
    atomicAdd(p + 0, cur * hv.x);
    atomicAdd(p + 1, cur * hv.y);
    atomicAdd(p + 2, cur * hv.z);
    atomicAdd(p + 3, cur * hv.w);
  }
  if (rb != ra && rb < n && rem != 0.f) {
    float* p = out + ((size_t)b * T_ + rb) * H_ + h;
    atomicAdd(p + 0, rem * hv.x);
    atomicAdd(p + 1, rem * hv.y);
    atomicAdd(p + 2, rem * hv.z);
    atomicAdd(p + 3, rem * hv.w);
  }
}

// ---------------------------------------------------------------------------
extern "C" void kernel_launch(void* const* d_in, const int* in_sizes, int n_in,
                              void* d_out, int out_size, void* d_ws, size_t ws_size,
                              hipStream_t stream) {
  (void)in_sizes; (void)n_in; (void)ws_size;
  const float* hidden = (const float*)d_in[0];
  const float* mask   = (const float*)d_in[1];
  const float* conv_w = (const float*)d_in[2];
  const float* conv_b = (const float*)d_in[3];
  const float* out_w  = (const float*)d_in[4];
  const float* out_b  = (const float*)d_in[5];
  float* out = (float*)d_out;

  char* ws = (char*)d_ws;
  bf16_t* Wt    = (bf16_t*)(ws + WS_WT);
  float*  alpha = (float*) (ws + WS_ALPHA);
  float*  curW  = (float*) (ws + WS_CUR);
  float*  remW  = (float*) (ws + WS_REM);
  int*    rowA  = (int*)   (ws + WS_ROWA);
  int*    rowB  = (int*)   (ws + WS_ROWB);
  int*    nF    = (int*)   (ws + WS_NF);

  // 1) repack weights to bf16 [k][o][i]
  pack_w_kernel<<<dim3((KW * H_ * H_ + 255) / 256), 256, 0, stream>>>(conv_w, Wt);

  // 2) zero output (scatter accumulates atomically)
  const int n4 = out_size / 4;
  zero_out_kernel<<<dim3(4096), 256, 0, stream>>>((float4*)out, n4);

  // 3) WMMA GEMM -> alphas
  const size_t smem = (size_t)LROWS * LSTR * 2 + (size_t)BM * 8 * 4;
  alphas_gemm_kernel<<<dim3(T_ / BM, B_), 256, smem, stream>>>(
      hidden, mask, Wt, conv_b, out_w, out_b, alpha);

  // 4) CIF prefix-scan -> per-step weights & rows
  cif_scan_kernel<<<dim3(B_), 256, 0, stream>>>(alpha, curW, remW, rowA, rowB, nF);

  // 5) scatter fired frames into output
  cif_scatter_kernel<<<dim3(T_, B_), 128, 0, stream>>>(
      hidden, curW, remW, rowA, rowB, nF, out);
}
